// EncoderProcesserDecoder_66924180406932
// MI455X (gfx1250) — matrix operand order repaired and checked by reading, compiled, not run on hardware
//
#include <hip/hip_runtime.h>

// ---------------------------------------------------------------------------
// MeshGraphNet-style encode-process-decode, CDNA5 (gfx1250, wave32, WMMA).
// Input order (from setup_inputs() + jax pytree flatten, dict keys sorted):
//   d_in[0] = x              f32 [50000,2]
//   d_in[1] = edge_attr      f32 [400000,3]
//   d_in[2] = edge_index     i32 [2,400000]  (row0 = senders, row1 = receivers)
//   d_in[3..] = param leaves: blocks[0..3]{edge,node,receiver,sender}, dec,
//               enc_edge, enc_node; each MLP = 4 x (W,b); leaf idx:
//               3 + mlp*8 + layer*2 (+1 for bias)
// ---------------------------------------------------------------------------

#define N_NODESC 50000
#define N_EDGESC 400000
#define HIDC 128

typedef __attribute__((ext_vector_type(16))) __bf16 v16bf;
typedef __attribute__((ext_vector_type(8)))  float  v8f;

// ---------------------------------------------------------------------------
// Weight prep: f32 [K,128] row-major -> bf16 WMMA B-fragment-ready layout:
// frag (kk = k/32, t = n/16); lane L = ((k%32)/16)*16 + (n%16) holds 16
// contiguous bf16, element j = k%16.  (contiguous k order, same as the A fragments)
// A 128-wide K chunk of this layout is one contiguous 32KB slice -> the GEMM
// kernel stages it into LDS with a straight b128 memcpy.
// ---------------------------------------------------------------------------
__global__ void prep_weight_kernel(const float* __restrict__ W,
                                   __bf16* __restrict__ Wp, int K) {
  int i = blockIdx.x * blockDim.x + threadIdx.x;
  if (i >= K * 128) return;
  int k = i >> 7, n = i & 127;
  int kk = k >> 5, rem = k & 31, halfk = rem >> 4, j = rem & 15;
  int t = n >> 4, nn = n & 15, L = halfk * 16 + nn;
  Wp[(((size_t)(kk * 8 + t) * 32 + L) << 4) + j] = (__bf16)W[(size_t)k * 128 + n];
}

// ---------------------------------------------------------------------------
// Core WMMA GEMM layer: Y[R,128] = f(X[R,K] @ W[K,128] + b)
// MODE 0: Y bf16 = relu(.)   MODE 1: Y bf16 = (.)
// MODE 2: Y f32  = (.)       MODE 3: Y f32 += (.)
// 256 threads = 8 waves; block tile 128x128; wave tile 16x128 (8 accum tiles).
// Per 128-wide K chunk: X tile (32KB, 16B chunks rotated by row: pc=(c+r)&15
// for bank-spread, 16B-aligned A reads) and W chunk (32KB fragment-ready) are
// both staged in LDS.  A and B fragments are software-pipelined one step deep
// so each WMMA overlaps with the next fragment's ds_loads (dscnt>0 waits).
// Epilogue transposes D through LDS -> coalesced b128/f32x4 global stores.
// Static LDS = exactly 64KB.
// ---------------------------------------------------------------------------
template <int MODE>
__global__ __launch_bounds__(256) void gemm_wmma_kernel(
    const __bf16* __restrict__ X, const __bf16* __restrict__ Wp,
    const float* __restrict__ bias, void* __restrict__ Yraw, int R, int K) {
  __shared__ __align__(16) unsigned char smem[65536];
  __bf16* xs   = (__bf16*)smem;            // 128 x 128 bf16 X tile (swizzled)
  __bf16* wsld = (__bf16*)(smem + 32768);  // 4x8x32x16 bf16 W fragments

  const int tid = threadIdx.x;
  const int wave = tid >> 5;
  const int lane = tid & 31;
  const int half = lane >> 4;
  const int m = lane & 15;
  const int rowBase = blockIdx.x * 128;
  const int lr = wave * 16 + m;  // this lane's local row within the tile

  union AF { v16bf v; uint4 q[2]; };
  union BF { v16bf v; uint4 q[2]; };

  v8f acc[8] = {};

  for (int kc = 0; kc < K; kc += 128) {
    __syncthreads();
    // ---- stage W chunk: contiguous 32KB slice of fragment-ready weights ----
    {
      const uint4* src =
          reinterpret_cast<const uint4*>(Wp + ((size_t)(kc >> 5) << 12));
      uint4* dst = reinterpret_cast<uint4*>(wsld);
      for (int i = tid; i < 2048; i += 256) dst[i] = src[i];
    }
    // ---- stage X chunk: 128 rows x 16 chunks of 8 bf16, row-rotated ----
    for (int i = tid; i < 2048; i += 256) {
      int r = i >> 4, c = i & 15;
      int gr = rowBase + r;
      if (gr >= R) gr = R - 1;  // clamp: tail rows replicate, stores guarded
      const uint4 v =
          *reinterpret_cast<const uint4*>(X + (size_t)gr * K + kc + c * 8);
      if (kc + 128 < K)  // prefetch next K chunk (K=256 node-MLP layer)
        __builtin_prefetch(X + (size_t)gr * K + kc + 128 + c * 8, 0, 1);
      int pc = (c + r) & 15;
      *reinterpret_cast<uint4*>(&xs[r * 128 + pc * 8]) = v;
    }
    __syncthreads();

    // A-fragment LDS chunk addresses (un-rotated) for a given kk:
    //   run1 chunk = kk*4 + 2*half, run2 = run1 + 1  (contiguous k = 16*half + s, matches B)
    auto loadA = [&](int kk, AF& A) {
      const int c0 = kk * 4 + half * 2;
      const int p0 = (c0 + lr) & 15;
      const int p1 = (c0 + 1 + lr) & 15;
      A.q[0] = *reinterpret_cast<const uint4*>(&xs[lr * 128 + p0 * 8]);
      A.q[1] = *reinterpret_cast<const uint4*>(&xs[lr * 128 + p1 * 8]);
    };

    AF Ac, An;
    loadA(0, Ac);
#pragma unroll
    for (int kk = 0; kk < 4; ++kk) {
      const __bf16* wb = &wsld[((kk * 8) * 32 + lane) << 4];
      BF Bc, Bn;
      Bc.q[0] = reinterpret_cast<const uint4*>(wb)[0];
      Bc.q[1] = reinterpret_cast<const uint4*>(wb)[1];
#pragma unroll
      for (int t = 0; t < 8; ++t) {
        if (t < 7) {  // prefetch next N-tile's B before consuming current
          const uint4* wq =
              reinterpret_cast<const uint4*>(wb + ((t + 1) * 32 << 4));
          Bn.q[0] = wq[0];
          Bn.q[1] = wq[1];
        } else if (kk < 3) {  // prefetch next kk's A before last WMMA
          loadA(kk + 1, An);
        }
        acc[t] = __builtin_amdgcn_wmma_f32_16x16x32_bf16(
            false, Ac.v, false, Bc.v, (short)0, acc[t], false, false);
        if (t < 7) Bc = Bn;
      }
      if (kk < 3) Ac = An;
    }
  }

  // ---- epilogue: D tile layout -> lane (half,n): VGPR v = (M=v+8*half, N=n).
  // Transpose through LDS, then coalesced global stores. ----
  const int n16 = lane & 15;
  __syncthreads();
  if (MODE == 0 || MODE == 1) {
    __bf16* ob = (__bf16*)smem;  // 128x128 bf16 output tile
#pragma unroll
    for (int t = 0; t < 8; ++t) {
      const float bv = bias[t * 16 + n16];
#pragma unroll
      for (int v = 0; v < 8; ++v) {
        float val = acc[t][v] + bv;
        if (MODE == 0) val = fmaxf(val, 0.0f);
        ob[(wave * 16 + half * 8 + v) * 128 + t * 16 + n16] = (__bf16)val;
      }
    }
    __syncthreads();
    __bf16* Y = (__bf16*)Yraw;
    for (int i = tid; i < 2048; i += 256) {
      int r = i >> 4, c = i & 15;
      int gr = rowBase + r;
      if (gr < R)
        *reinterpret_cast<uint4*>(Y + (size_t)gr * 128 + c * 8) =
            *reinterpret_cast<const uint4*>(&ob[r * 128 + c * 8]);
    }
  } else {
    float* of = (float*)smem;  // 128x128 f32 output tile (uses full 64KB)
#pragma unroll
    for (int t = 0; t < 8; ++t) {
      const float bv = bias[t * 16 + n16];
#pragma unroll
      for (int v = 0; v < 8; ++v)
        of[(wave * 16 + half * 8 + v) * 128 + t * 16 + n16] = acc[t][v] + bv;
    }
    __syncthreads();
    float* Y = (float*)Yraw;
    for (int i = tid; i < 4096; i += 256) {
      int r = i >> 5, c = i & 31;
      int gr = rowBase + r;
      if (gr < R) {
        float4* gp = reinterpret_cast<float4*>(Y + (size_t)gr * 128 + c * 4);
        float4 lv = *reinterpret_cast<const float4*>(&of[r * 128 + c * 4]);
        if (MODE == 3) {
          float4 g = *gp;
          lv.x += g.x; lv.y += g.y; lv.z += g.z; lv.w += g.w;
        }
        *gp = lv;
      }
    }
  }
}

// ---------------------------------------------------------------------------
// Small helper kernels
// ---------------------------------------------------------------------------
__global__ void mlp_in_small_kernel(const float* __restrict__ x,
                                    const float* __restrict__ W,
                                    const float* __restrict__ b,
                                    __bf16* __restrict__ y, int R, int K) {
  int i = blockIdx.x * blockDim.x + threadIdx.x;
  if (i >= R * 128) return;
  int r = i >> 7, c = i & 127;
  float acc = b[c];
  for (int k = 0; k < K; ++k) acc += x[(size_t)r * K + k] * W[k * 128 + c];
  y[i] = (__bf16)fmaxf(acc, 0.0f);
}

__global__ void dec_final_kernel(const __bf16* __restrict__ x,
                                 const float* __restrict__ W,
                                 const float* __restrict__ b,
                                 float* __restrict__ out, int R) {
  int i = blockIdx.x * blockDim.x + threadIdx.x;
  if (i >= R * 2) return;
  int r = i >> 1, c = i & 1;
  float acc = b[c];
#pragma unroll 8
  for (int k = 0; k < 128; ++k)
    acc += (float)x[(size_t)r * 128 + k] * W[k * 2 + c];
  out[i] = acc;
}

__global__ void gather_kernel(const __bf16* __restrict__ src,
                              const int* __restrict__ idx,
                              __bf16* __restrict__ dst, int E) {
  int i = blockIdx.x * blockDim.x + threadIdx.x;  // E * 16 chunks of 16B
  if (i >= E * 16) return;
  int e = i >> 4, c = i & 15;
  int s = idx[e];
  reinterpret_cast<uint4*>(dst)[(size_t)e * 16 + c] =
      reinterpret_cast<const uint4*>(src)[(size_t)s * 16 + c];
}

__global__ void zero_kernel(float* __restrict__ p, int n) {
  int i = blockIdx.x * blockDim.x + threadIdx.x;
  if (i < n) p[i] = 0.0f;
}

__global__ void scatter_add_kernel(const float* __restrict__ e_new,
                                   const int* __restrict__ recv,
                                   float* __restrict__ agg, int E) {
  int i = blockIdx.x * blockDim.x + threadIdx.x;
  if (i >= E * 128) return;
  int e = i >> 7, c = i & 127;
  unsafeAtomicAdd(&agg[(size_t)recv[e] * 128 + c], e_new[i]);
}

__global__ void concat_kernel(const __bf16* __restrict__ node,
                              const float* __restrict__ agg,
                              __bf16* __restrict__ cat, int N) {
  int i = blockIdx.x * blockDim.x + threadIdx.x;
  if (i >= N * 256) return;
  int r = i >> 8, c = i & 255;
  cat[i] = (c < 128) ? node[(size_t)r * 128 + c]
                     : (__bf16)agg[(size_t)r * 128 + (c - 128)];
}

__global__ void residual_kernel(__bf16* __restrict__ x,
                                const float* __restrict__ delta, size_t n) {
  size_t i = (size_t)blockIdx.x * blockDim.x + threadIdx.x;
  if (i >= n) return;
  x[i] = (__bf16)((float)x[i] + delta[i]);
}

// ---------------------------------------------------------------------------
// Orchestration
// ---------------------------------------------------------------------------
extern "C" void kernel_launch(void* const* d_in, const int* in_sizes, int n_in,
                              void* d_out, int out_size, void* d_ws,
                              size_t ws_size, hipStream_t stream) {
  (void)in_sizes; (void)n_in; (void)out_size; (void)ws_size;

  const float* x         = (const float*)d_in[0];
  const float* edge_attr = (const float*)d_in[1];
  const int*   senders   = (const int*)d_in[2];
  const int*   receivers = senders + N_EDGESC;

  auto leafW = [&](int mlp, int l) { return (const float*)d_in[3 + mlp * 8 + l * 2]; };
  auto leafB = [&](int mlp, int l) { return (const float*)d_in[3 + mlp * 8 + l * 2 + 1]; };

  char* ws = (char*)d_ws;
  size_t off = 0;
  auto carve = [&](size_t bytes) -> char* {
    char* p = ws + off;
    off += (bytes + 255) & ~(size_t)255;
    return p;
  };

  __bf16* Wp      = (__bf16*)carve(1261568ull * 2);               // prepped weights
  __bf16* node_bf = (__bf16*)carve((size_t)N_NODESC * HIDC * 2);
  __bf16* edge_bf = (__bf16*)carve((size_t)N_EDGESC * HIDC * 2);
  __bf16* tA      = (__bf16*)carve((size_t)N_EDGESC * HIDC * 2);  // ping
  __bf16* tB      = (__bf16*)carve((size_t)N_EDGESC * HIDC * 2);  // pong
  float*  e_new   = (float*)carve((size_t)N_EDGESC * HIDC * 4);
  float*  agg     = (float*)carve((size_t)N_NODESC * HIDC * 4);
  float*  nnew    = (float*)carve((size_t)N_NODESC * HIDC * 4);
  __bf16* catb    = (__bf16*)carve((size_t)N_NODESC * 2 * HIDC * 2);

  // MLP index map: blocks[b]: edge=b*4+0, node=b*4+1, receiver=b*4+2,
  // sender=b*4+3; dec=16; enc_edge=17; enc_node=18.
  const int MLP_DEC = 16, MLP_EE = 17, MLP_EN = 18;

  // ---- weight prep (fragment-ready bf16) ----
  size_t wpoff[19][4];
  size_t cur = 0;
  for (int mlp = 0; mlp < 19; ++mlp) {
    for (int l = 0; l < 4; ++l) {
      bool skip = (mlp == MLP_EN && l == 0) || (mlp == MLP_EE && l == 0) ||
                  (mlp == MLP_DEC && l == 3);  // tiny-K layers: raw f32
      if (skip) { wpoff[mlp][l] = 0; continue; }
      int K = (mlp < 16 && (mlp & 3) == 1 && l == 0) ? 256 : 128;
      int n = K * 128;
      prep_weight_kernel<<<(n + 255) / 256, 256, 0, stream>>>(
          leafW(mlp, l), Wp + cur, K);
      wpoff[mlp][l] = cur;
      cur += (size_t)n;
    }
  }

  auto gemm = [&](int mode, const __bf16* X, int mlp, int l, void* Y, int R,
                  int K) {
    dim3 grid((R + 127) / 128);
    const __bf16* w = Wp + wpoff[mlp][l];
    const float* b = leafB(mlp, l);
    if (mode == 0)
      gemm_wmma_kernel<0><<<grid, 256, 0, stream>>>(X, w, b, Y, R, K);
    else if (mode == 1)
      gemm_wmma_kernel<1><<<grid, 256, 0, stream>>>(X, w, b, Y, R, K);
    else if (mode == 2)
      gemm_wmma_kernel<2><<<grid, 256, 0, stream>>>(X, w, b, Y, R, K);
    else
      gemm_wmma_kernel<3><<<grid, 256, 0, stream>>>(X, w, b, Y, R, K);
  };

  // ---- Encoder: node ----
  mlp_in_small_kernel<<<(N_NODESC * HIDC + 255) / 256, 256, 0, stream>>>(
      x, leafW(MLP_EN, 0), leafB(MLP_EN, 0), tA, N_NODESC, 2);
  gemm(0, tA, MLP_EN, 1, tB, N_NODESC, 128);
  gemm(0, tB, MLP_EN, 2, tA, N_NODESC, 128);
  gemm(1, tA, MLP_EN, 3, node_bf, N_NODESC, 128);  // last layer: no relu

  // ---- Encoder: edge ----
  mlp_in_small_kernel<<<(N_EDGESC * HIDC + 255) / 256, 256, 0, stream>>>(
      edge_attr, leafW(MLP_EE, 0), leafB(MLP_EE, 0), tA, N_EDGESC, 3);
  gemm(0, tA, MLP_EE, 1, tB, N_EDGESC, 128);
  gemm(0, tB, MLP_EE, 2, tA, N_EDGESC, 128);
  gemm(1, tA, MLP_EE, 3, edge_bf, N_EDGESC, 128);

  // ---- Processor: 4 GnBlocks ----
  for (int blk = 0; blk < 4; ++blk) {
    const int M_EDGE = blk * 4 + 0, M_NODE = blk * 4 + 1;
    const int M_RECV = blk * 4 + 2, M_SEND = blk * 4 + 3;

    // sender MLP on gathered sender node features -> e_new (overwrite)
    gather_kernel<<<(N_EDGESC * 16 + 255) / 256, 256, 0, stream>>>(
        node_bf, senders, tA, N_EDGESC);
    gemm(0, tA, M_SEND, 0, tB, N_EDGESC, 128);
    gemm(0, tB, M_SEND, 1, tA, N_EDGESC, 128);
    gemm(0, tA, M_SEND, 2, tB, N_EDGESC, 128);
    gemm(2, tB, M_SEND, 3, e_new, N_EDGESC, 128);

    // receiver MLP -> e_new (accumulate)
    gather_kernel<<<(N_EDGESC * 16 + 255) / 256, 256, 0, stream>>>(
        node_bf, receivers, tA, N_EDGESC);
    gemm(0, tA, M_RECV, 0, tB, N_EDGESC, 128);
    gemm(0, tB, M_RECV, 1, tA, N_EDGESC, 128);
    gemm(0, tA, M_RECV, 2, tB, N_EDGESC, 128);
    gemm(3, tB, M_RECV, 3, e_new, N_EDGESC, 128);

    // edge MLP -> e_new (accumulate)
    gemm(0, edge_bf, M_EDGE, 0, tA, N_EDGESC, 128);
    gemm(0, tA, M_EDGE, 1, tB, N_EDGESC, 128);
    gemm(0, tB, M_EDGE, 2, tA, N_EDGESC, 128);
    gemm(3, tA, M_EDGE, 3, e_new, N_EDGESC, 128);

    // scatter-sum e_new over receivers -> agg
    zero_kernel<<<(N_NODESC * HIDC + 255) / 256, 256, 0, stream>>>(
        agg, N_NODESC * HIDC);
    scatter_add_kernel<<<(N_EDGESC * HIDC + 255) / 256, 256, 0, stream>>>(
        e_new, receivers, agg, N_EDGESC);

    // node MLP on concat(node, agg) -> nnew
    concat_kernel<<<(N_NODESC * 2 * HIDC + 255) / 256, 256, 0, stream>>>(
        node_bf, agg, catb, N_NODESC);
    gemm(0, catb, M_NODE, 0, tA, N_NODESC, 256);
    gemm(0, tA, M_NODE, 1, tB, N_NODESC, 128);
    gemm(0, tB, M_NODE, 2, tA, N_NODESC, 128);
    gemm(2, tA, M_NODE, 3, nnew, N_NODESC, 128);

    // residuals
    residual_kernel<<<(N_NODESC * HIDC + 255) / 256, 256, 0, stream>>>(
        node_bf, nnew, (size_t)N_NODESC * HIDC);
    residual_kernel<<<((int)(((size_t)N_EDGESC * HIDC + 255) / 256)), 256, 0,
                      stream>>>(edge_bf, e_new, (size_t)N_EDGESC * HIDC);
  }

  // ---- Decoder ----
  gemm(0, node_bf, MLP_DEC, 0, tA, N_NODESC, 128);
  gemm(0, tA, MLP_DEC, 1, tB, N_NODESC, 128);
  gemm(0, tB, MLP_DEC, 2, tA, N_NODESC, 128);
  dec_final_kernel<<<(N_NODESC * 2 + 255) / 256, 256, 0, stream>>>(
      tA, leafW(MLP_DEC, 3), leafB(MLP_DEC, 3), (float*)d_out, N_NODESC);
}
